// SE3TransformerLayer_7275674599858
// MI455X (gfx1250) — matrix operand order repaired and checked by reading, not compiled, not run on hardware
//
#include <hip/hip_runtime.h>
#include <hip/hip_bf16.h>

// ---------------------------------------------------------------------------
// SE(3)-transformer layer for MI455X (gfx1250, wave32).
// Dense GEMMs -> bf16 WMMA (v_wmma_f32_16x16x32_bf16); A-tile staged by the
// Tensor Data Mover (TENSOR_LOAD_TO_LDS, TENSORcnt) when the builtin exists,
// else per-lane async-to-LDS (ASYNCcnt), else manual VMEM+DS path.
// Edge phase  -> wave-per-edge gathers + butterfly reductions + f32 atomics
//               (working set fits the 192MB L2, so atomics stay on-chip).
// ---------------------------------------------------------------------------

#define Nn  10000
#define Ee  160000
#define Ss  512
#define Vv  128
#define Hh  8
#define HDd 64
#define Rr  32
#define CUTOFF 10.0f
#define PI_F 3.14159265358979323846f

typedef __bf16 bf16;
typedef __attribute__((ext_vector_type(16))) __bf16 v16bf;
typedef __attribute__((ext_vector_type(8)))  float  v8f;
typedef __attribute__((ext_vector_type(4)))  unsigned u32x4;
typedef __attribute__((ext_vector_type(8)))  int      i32x8;
typedef __attribute__((ext_vector_type(4)))  int      i32x4;

#if __has_builtin(__builtin_amdgcn_tensor_load_to_lds) && \
    __has_builtin(__builtin_amdgcn_s_wait_tensorcnt)
#define USE_TDM_A 1
#endif
#if __has_builtin(__builtin_amdgcn_global_load_async_to_lds_b128)
#define USE_ASYNC_A 1
#endif

// -------------------------- small helpers ----------------------------------

__device__ __forceinline__ bf16 f2bf(float f) {
  unsigned u = __float_as_uint(f);
  unsigned r = (u + 0x7fffu + ((u >> 16) & 1u)) >> 16;
  return __builtin_bit_cast(bf16, (unsigned short)r);
}

// monotonic float<->unsigned encoding for atomicMax-based segment max
__device__ __forceinline__ unsigned fenc(float f) {
  unsigned u = __float_as_uint(f);
  return (u & 0x80000000u) ? ~u : (u | 0x80000000u);
}
__device__ __forceinline__ float fdec(unsigned u) {
  return (u & 0x80000000u) ? __uint_as_float(u & 0x7fffffffu)
                           : __uint_as_float(~u);
}

__device__ __forceinline__ float wave_allreduce(float v) {
#pragma unroll
  for (int m = 16; m >= 1; m >>= 1) v += __shfl_xor(v, m, 32);
  return v;
}

// -------------------------- cast / pack kernels ----------------------------

__global__ void k_cast_bf16(const float* __restrict__ src,
                            bf16* __restrict__ dst, int n) {
  int i = blockIdx.x * blockDim.x + threadIdx.x;
  if (i < n) dst[i] = f2bf(src[i]);
}

// vectors (N,V,3) f32 -> 3 coord-major bf16 planes X[i][n][v]
__global__ void k_split_vec(const float* __restrict__ vec,
                            bf16* __restrict__ X) {
  int idx = blockIdx.x * blockDim.x + threadIdx.x;
  if (idx >= Nn * Vv * 3) return;
  int i = idx % 3;
  int v = (idx / 3) % Vv;
  int n = idx / (3 * Vv);
  X[((size_t)i * Nn + n) * Vv + v] = f2bf(vec[idx]);
}

// d_out[N*S + n*V*3 + v*3 + i] = vectors + out_v_fin (coord-major planes)
__global__ void k_add_vec(const float* __restrict__ vec,
                          const float* __restrict__ planes,
                          float* __restrict__ out) {
  int idx = blockIdx.x * blockDim.x + threadIdx.x;
  if (idx >= Nn * Vv * 3) return;
  int i = idx % 3;
  int v = (idx / 3) % Vv;
  int n = idx / (3 * Vv);
  out[idx] = vec[idx] + planes[((size_t)i * Nn + n) * Vv + v];
}

// -------------------------- bf16 WMMA GEMM ---------------------------------
// C[M x Nc] = A[M x K](bf16,row-major) * B[K x Nc](bf16,row-major) + bias
// Tile: 64x64x32. Block = 128 threads (4 waves); wave w owns rows w*16..+16.
// A tile staged via TDM (one TENSOR_LOAD_TO_LDS per K-step, issued by wave 0,
// completion on TENSORcnt) when available. B staged transposed in LDS so
// per-lane fragment loads are contiguous 32-bit LDS reads.

__global__ __launch_bounds__(128)
void k_gemm_bf16(const bf16* __restrict__ A, const bf16* __restrict__ B,
                 const float* __restrict__ bias, float* __restrict__ C,
                 int M, int K, int Nc) {
  __shared__ __align__(16) bf16 As[64][32];
  __shared__ __align__(16) bf16 Bt[64][32];  // Bt[n][k]

  const int tid  = threadIdx.x;
  const int wave = tid >> 5;
  const int lane = tid & 31;
  const int half = lane >> 4;
  const int lm   = lane & 15;

  const int m0 = blockIdx.y * 64;
  const int n0 = blockIdx.x * 64;

  v8f acc[4];
#pragma unroll
  for (int i = 0; i < 4; ++i) acc[i] = (v8f){};

  // A loader: thread t covers (row = t/2, 16 cols at (t&1)*16)
  const int ar  = tid >> 1;
  const int as  = (tid & 1) * 16;
  const int agm = (m0 + ar < M) ? (m0 + ar) : (M - 1);
  // B loader: thread t covers (k = t/4, 16 cols at (t&3)*16)
  const int bk  = tid >> 2;
  const int bns = (tid & 3) * 16;

#if defined(USE_TDM_A)
  const unsigned lds_a = (unsigned)(size_t)(void*)&As[0][0];
#endif

  for (int k0 = 0; k0 < K; k0 += 32) {
    // ---- B tile: per-lane global read (coalesced rows of B) ----
    const uint4* bp =
        reinterpret_cast<const uint4*>(B + (size_t)(k0 + bk) * Nc + n0 + bns);
    uint4 b0 = bp[0], b1 = bp[1];

#if !defined(USE_TDM_A) && !defined(USE_ASYNC_A)
    const uint4* ap =
        reinterpret_cast<const uint4*>(A + (size_t)agm * K + k0 + as);
    uint4 a0 = ap[0], a1 = ap[1];
#endif

    if (k0 + 32 < K) {  // prefetch next K-slab of B (global_prefetch_b8)
      __builtin_prefetch(B + (size_t)(k0 + 32 + bk) * Nc + n0 + bns, 0, 1);
    }

    __syncthreads();  // previous iteration's LDS reads are done

#if defined(USE_TDM_A)
    // ---- A tile via Tensor Data Mover: 64x32 bf16 2D tile, row stride K ----
    if (wave == 0) {
      unsigned long long ga =
          (unsigned long long)(size_t)(const void*)(A + (size_t)m0 * K + k0);
      u32x4 g0;
      g0[0] = 1u;                                   // count=1, user mode
      g0[1] = lds_a;                                // lds_addr (bytes)
      g0[2] = (unsigned)(ga & 0xffffffffu);         // global_addr[31:0]
      g0[3] = (unsigned)((ga >> 32) & 0x01ffffffu)  // global_addr[56:32]
              | (2u << 30);                         // type=2 ("image")
      i32x8 g1;
      g1[0] = (int)(1u << 16);                      // data_size=1 -> 2 bytes
      g1[1] = (int)((unsigned)(K & 0xffff) << 16);  // tensor_dim0[15:0]
      g1[2] = (int)(((unsigned)K >> 16) | ((unsigned)(M & 0xffff) << 16));
      g1[3] = (int)(((unsigned)M >> 16) | (32u << 16));  // tile_dim0=32
      g1[4] = (int)64u;                             // tile_dim1=64, tile_dim2=0
      g1[5] = (int)(unsigned)K;                     // tensor_dim0_stride = K
      g1[6] = 0;
      g1[7] = 0;
      i32x4 g2 = {0, 0, 0, 0};
      i32x4 g3 = {0, 0, 0, 0};
      i32x8 g4 = {0, 0, 0, 0, 0, 0, 0, 0};
      __builtin_amdgcn_tensor_load_to_lds(g0, g1, g2, g3, g4, 0);
      __builtin_amdgcn_s_wait_tensorcnt(0);
    }
#elif defined(USE_ASYNC_A)
    // ---- A tile via per-lane async copy to LDS (ASYNCcnt) ----
    __builtin_amdgcn_global_load_async_to_lds_b128(
        (__attribute__((address_space(1))) void*)(void*)(A + (size_t)agm * K +
                                                         k0 + as),
        (__attribute__((address_space(3))) void*)(void*)&As[ar][as], 0, 0);
    __builtin_amdgcn_global_load_async_to_lds_b128(
        (__attribute__((address_space(1))) void*)(void*)(A + (size_t)agm * K +
                                                         k0 + as + 8),
        (__attribute__((address_space(3))) void*)(void*)&As[ar][as + 8], 0, 0);
#if __has_builtin(__builtin_amdgcn_s_wait_asynccnt)
    __builtin_amdgcn_s_wait_asynccnt(0);
#else
    asm volatile("s_wait_asynccnt 0" ::: "memory");
#endif
#else
    *reinterpret_cast<uint4*>(&As[ar][as])     = a0;
    *reinterpret_cast<uint4*>(&As[ar][as + 8]) = a1;
#endif

    // B transpose into LDS (scalar 16-bit scatters)
    bf16 tmp[16];
    *reinterpret_cast<uint4*>(tmp)     = b0;
    *reinterpret_cast<uint4*>(tmp + 8) = b1;
#pragma unroll
    for (int j = 0; j < 16; ++j) Bt[bns + j][bk] = tmp[j];
    __syncthreads();

    // A fragment: lane(half,lm): K(v) = (v>=4?16:0) + half*8 + (v&3)*2
    v16bf afrag;
    const bf16* arow = &As[wave * 16 + lm][0];
#pragma unroll
    for (int v = 0; v < 8; ++v) {
      int kb = ((v & 4) ? 16 : 0) + half * 8 + (v & 3) * 2;
      afrag[2 * v]     = arow[kb];
      afrag[2 * v + 1] = arow[kb + 1];
    }

    // load all four B fragments first so the ds waits hoist, then issue the
    // four WMMAs back-to-back
    v16bf bfrag[4];
#pragma unroll
    for (int nt = 0; nt < 4; ++nt) {
      const bf16* bcol = &Bt[nt * 16 + lm][0];
#pragma unroll
      for (int v = 0; v < 8; ++v) {
        int kb = ((v & 4) ? 16 : 0) + half * 8 + (v & 3) * 2;
        bfrag[nt][2 * v]     = bcol[kb];
        bfrag[nt][2 * v + 1] = bcol[kb + 1];
      }
    }
#pragma unroll
    for (int nt = 0; nt < 4; ++nt) {
      acc[nt] = __builtin_amdgcn_wmma_f32_16x16x32_bf16(
          false, afrag, false, bfrag[nt], (short)0, acc[nt], false, false);
    }
  }

  // C layout: VGPR r -> M = half*8 + r, lane lm -> N
#pragma unroll
  for (int nt = 0; nt < 4; ++nt) {
    int n = n0 + nt * 16 + lm;
    float bv = bias ? bias[n] : 0.0f;
#pragma unroll
    for (int r = 0; r < 8; ++r) {
      int m = m0 + wave * 16 + half * 8 + r;
      if (m < M) C[(size_t)m * Nc + n] = acc[nt][r] + bv;
    }
  }
}

// -------------------------- edge-phase kernels -----------------------------

// one wave32 per edge: attention logits (scalar dots + vector dots + Bessel
// radial MLP), fused with atomicMax segment-max update.
__global__ __launch_bounds__(256)
void k_edge_logits(const long long* __restrict__ ei,
                   const float* __restrict__ qs, const float* __restrict__ ks,
                   const float* __restrict__ qv, const float* __restrict__ kv,
                   const float* __restrict__ evec,
                   const float* __restrict__ W_edge,
                   const float* __restrict__ b_edge,
                   float* __restrict__ logits, unsigned* __restrict__ maxbuf) {
  int e = blockIdx.x * 8 + (threadIdx.x >> 5);
  if (e >= Ee) return;
  int lane = threadIdx.x & 31;
  int row = (int)ei[e];
  int col = (int)ei[Ee + e];

  const float* qrow = qs + (size_t)row * Ss;
  const float* kcol = ks + (size_t)col * Ss;

  float att_s[Hh];
#pragma unroll
  for (int h = 0; h < Hh; ++h) {
    int b = h * HDd + lane;
    float p = qrow[b] * kcol[b] + qrow[b + 32] * kcol[b + 32];
    att_s[h] = wave_allreduce(p);
  }

  float pv = 0.0f;
#pragma unroll
  for (int i = 0; i < 3; ++i) {
    const float* q = qv + ((size_t)i * Nn + row) * Vv;
    const float* k = kv + ((size_t)i * Nn + col) * Vv;
#pragma unroll
    for (int w = 0; w < Vv; w += 32) pv += q[w + lane] * k[w + lane];
  }
  float att_v = wave_allreduce(pv);

  // radial basis: lane r in [0,32) computes Bessel term r
  float ex = evec[3 * e], ey = evec[3 * e + 1], ez = evec[3 * e + 2];
  float d = sqrtf(ex * ex + ey * ey + ez * ez);
  float freq = (float)(lane + 1) * (PI_F / CUTOFF);
  float cut  = 0.5f * (cosf(PI_F * d / CUTOFF) + 1.0f) * (d < CUTOFF ? 1.0f : 0.0f);
  float bas  = sinf(freq * d) / fmaxf(d, 1e-8f) * cut;

  const float inv_sqrt3 = 0.57735026918962576f;
  const float inv_sqrt_hd = 0.125f;  // 1/sqrt(64)
#pragma unroll
  for (int h = 0; h < Hh; ++h) {
    float ea = wave_allreduce(bas * W_edge[lane * Hh + h]) + b_edge[h];
    float lg = (att_s[h] + att_v * inv_sqrt3 + ea) * inv_sqrt_hd;
    if (lane == 0) {
      logits[(size_t)e * Hh + h] = lg;
      atomicMax(&maxbuf[row * Hh + h], fenc(lg));
    }
  }
}

// one thread per (edge, head): e = exp(l - m); accumulate segment sums
__global__ void k_edge_expsum(const long long* __restrict__ ei,
                              float* __restrict__ logits,
                              const unsigned* __restrict__ maxbuf,
                              float* __restrict__ sumbuf) {
  int idx = blockIdx.x * blockDim.x + threadIdx.x;
  if (idx >= Ee * Hh) return;
  int e = idx >> 3;  // Hh == 8
  int h = idx & 7;
  int row = (int)ei[e];
  float m = fdec(maxbuf[row * Hh + h]);
  float ex = expf(logits[idx] - m);
  logits[idx] = ex;  // in-place: now holds exp-values
  atomicAdd(&sumbuf[row * Hh + h], ex);
}

// one wave32 per edge: normalize attention, scatter msg_s and msg_v
__global__ __launch_bounds__(256)
void k_edge_scatter(const long long* __restrict__ ei,
                    const float* __restrict__ expv,
                    const float* __restrict__ sumbuf,
                    const float* __restrict__ vs, const float* __restrict__ vv,
                    float* __restrict__ out_s_agg,
                    float* __restrict__ out_v_agg) {
  int e = blockIdx.x * 8 + (threadIdx.x >> 5);
  if (e >= Ee) return;
  int lane = threadIdx.x & 31;
  int row = (int)ei[e];
  int col = (int)ei[Ee + e];

  float attn[Hh];
  float a_v = 0.0f;
#pragma unroll
  for (int h = 0; h < Hh; ++h) {
    attn[h] = expv[(size_t)e * Hh + h] / (sumbuf[row * Hh + h] + 1e-9f);
    a_v += attn[h];
  }
  a_v *= (1.0f / Hh);

  const float* vcol = vs + (size_t)col * Ss;
  float* orow = out_s_agg + (size_t)row * Ss;
#pragma unroll
  for (int h = 0; h < Hh; ++h) {
    int b = h * HDd + lane;
    atomicAdd(&orow[b],      attn[h] * vcol[b]);
    atomicAdd(&orow[b + 32], attn[h] * vcol[b + 32]);
  }
#pragma unroll
  for (int i = 0; i < 3; ++i) {
    const float* src = vv + ((size_t)i * Nn + col) * Vv;
    float* dst = out_v_agg + ((size_t)i * Nn + row) * Vv;
#pragma unroll
    for (int w = 0; w < Vv; w += 32)
      atomicAdd(&dst[w + lane], a_v * src[w + lane]);
  }
}

// fused residual + LayerNorm, one wave32 per node (16 elems/lane)
__global__ __launch_bounds__(256)
void k_layernorm(const float* __restrict__ x0, const float* __restrict__ dx,
                 const float* __restrict__ g, const float* __restrict__ b,
                 float* __restrict__ out) {
  int n = blockIdx.x * 8 + (threadIdx.x >> 5);
  if (n >= Nn) return;
  int lane = threadIdx.x & 31;
  const float* a = x0 + (size_t)n * Ss;
  const float* d = dx + (size_t)n * Ss;
  float x[16];
  float s = 0.0f;
#pragma unroll
  for (int j = 0; j < 16; ++j) {
    x[j] = a[lane + 32 * j] + d[lane + 32 * j];
    s += x[j];
  }
  s = wave_allreduce(s);
  float mu = s * (1.0f / Ss);
  float var = 0.0f;
#pragma unroll
  for (int j = 0; j < 16; ++j) {
    float t = x[j] - mu;
    var += t * t;
  }
  var = wave_allreduce(var);
  float rinv = rsqrtf(var * (1.0f / Ss) + 1e-5f);
#pragma unroll
  for (int j = 0; j < 16; ++j) {
    int c = lane + 32 * j;
    out[(size_t)n * Ss + c] = (x[j] - mu) * rinv * g[c] + b[c];
  }
}

// -------------------------- host orchestration -----------------------------

extern "C" void kernel_launch(void* const* d_in, const int* in_sizes, int n_in,
                              void* d_out, int out_size, void* d_ws,
                              size_t ws_size, hipStream_t stream) {
  (void)in_sizes; (void)n_in; (void)out_size; (void)ws_size;

  const float* scalars  = (const float*)d_in[0];
  const float* vectors  = (const float*)d_in[1];
  const float* edge_vec = (const float*)d_in[2];
  const float* Wq_s   = (const float*)d_in[3];
  const float* bq_s   = (const float*)d_in[4];
  const float* Wk_s   = (const float*)d_in[5];
  const float* bk_s   = (const float*)d_in[6];
  const float* Wv_s   = (const float*)d_in[7];
  const float* bv_s   = (const float*)d_in[8];
  const float* Wq_v   = (const float*)d_in[9];
  const float* Wk_v   = (const float*)d_in[10];
  const float* Wv_v   = (const float*)d_in[11];
  const float* Wout_s = (const float*)d_in[12];
  const float* bout_s = (const float*)d_in[13];
  const float* Wout_v = (const float*)d_in[14];
  const float* bout_v = (const float*)d_in[15];
  const float* W_edge = (const float*)d_in[16];
  const float* b_edge = (const float*)d_in[17];
  const float* ln_g   = (const float*)d_in[18];
  const float* ln_b   = (const float*)d_in[19];
  const long long* edge_index = (const long long*)d_in[20];

  // ---- workspace layout (256B aligned slabs) ----
  char* wptr = (char*)d_ws;
  auto alloc = [&](size_t bytes) -> void* {
    void* p = (void*)wptr;
    wptr += (bytes + 255) & ~(size_t)255;
    return p;
  };
  bf16* sc_bf  = (bf16*)alloc((size_t)Nn * Ss * 2);
  bf16* Wqs_bf = (bf16*)alloc((size_t)Ss * Ss * 2);
  bf16* Wks_bf = (bf16*)alloc((size_t)Ss * Ss * 2);
  bf16* Wvs_bf = (bf16*)alloc((size_t)Ss * Ss * 2);
  bf16* Wos_bf = (bf16*)alloc((size_t)Ss * Ss * 2);
  bf16* Wqv_bf = (bf16*)alloc((size_t)Vv * Vv * 2);
  bf16* Wkv_bf = (bf16*)alloc((size_t)Vv * Vv * 2);
  bf16* Wvv_bf = (bf16*)alloc((size_t)Vv * Vv * 2);
  bf16* Wov_bf = (bf16*)alloc((size_t)Vv * Vv * 2);
  bf16* Xv_bf  = (bf16*)alloc((size_t)3 * Nn * Vv * 2);  // coord-major planes
  float* q_s = (float*)alloc((size_t)Nn * Ss * 4);
  float* k_s = (float*)alloc((size_t)Nn * Ss * 4);
  float* v_s = (float*)alloc((size_t)Nn * Ss * 4);
  float* q_v = (float*)alloc((size_t)3 * Nn * Vv * 4);
  float* k_v = (float*)alloc((size_t)3 * Nn * Vv * 4);
  float* v_v = (float*)alloc((size_t)3 * Nn * Vv * 4);
  float* logits = (float*)alloc((size_t)Ee * Hh * 4);
  unsigned* maxbuf = (unsigned*)alloc((size_t)Nn * Hh * 4);
  float* sumbuf = (float*)alloc((size_t)Nn * Hh * 4);
  float* out_s_agg = (float*)alloc((size_t)Nn * Ss * 4);
  float* out_v_agg = (float*)alloc((size_t)3 * Nn * Vv * 4);
  bf16* out_s_bf = (bf16*)alloc((size_t)Nn * Ss * 2);
  bf16* out_v_bf = (bf16*)alloc((size_t)3 * Nn * Vv * 2);
  float* out_s_fin = (float*)alloc((size_t)Nn * Ss * 4);
  float* out_v_fin = (float*)alloc((size_t)3 * Nn * Vv * 4);

  auto cast = [&](const float* s, bf16* dstp, int n) {
    k_cast_bf16<<<(n + 255) / 256, 256, 0, stream>>>(s, dstp, n);
  };
  auto gemm = [&](const bf16* A, const bf16* B, const float* bias, float* C,
                  int M, int K, int Ncols) {
    dim3 grid(Ncols / 64, (M + 63) / 64);
    k_gemm_bf16<<<grid, 128, 0, stream>>>(A, B, bias, C, M, K, Ncols);
  };

  // 1) precision conversion
  cast(scalars, sc_bf, Nn * Ss);
  cast(Wq_s, Wqs_bf, Ss * Ss);
  cast(Wk_s, Wks_bf, Ss * Ss);
  cast(Wv_s, Wvs_bf, Ss * Ss);
  cast(Wout_s, Wos_bf, Ss * Ss);
  cast(Wq_v, Wqv_bf, Vv * Vv);
  cast(Wk_v, Wkv_bf, Vv * Vv);
  cast(Wv_v, Wvv_bf, Vv * Vv);
  cast(Wout_v, Wov_bf, Vv * Vv);
  k_split_vec<<<(Nn * Vv * 3 + 255) / 256, 256, 0, stream>>>(vectors, Xv_bf);

  // 2) node-level projections (WMMA)
  gemm(sc_bf, Wqs_bf, bq_s, q_s, Nn, Ss, Ss);
  gemm(sc_bf, Wks_bf, bk_s, k_s, Nn, Ss, Ss);
  gemm(sc_bf, Wvs_bf, bv_s, v_s, Nn, Ss, Ss);
  for (int i = 0; i < 3; ++i) {
    const bf16* Xi = Xv_bf + (size_t)i * Nn * Vv;
    gemm(Xi, Wqv_bf, nullptr, q_v + (size_t)i * Nn * Vv, Nn, Vv, Vv);
    gemm(Xi, Wkv_bf, nullptr, k_v + (size_t)i * Nn * Vv, Nn, Vv, Vv);
    gemm(Xi, Wvv_bf, nullptr, v_v + (size_t)i * Nn * Vv, Nn, Vv, Vv);
  }

  // 3) edge phase: logits -> segment softmax -> scatter
  (void)hipMemsetAsync(maxbuf, 0, (size_t)Nn * Hh * 4, stream);
  (void)hipMemsetAsync(sumbuf, 0, (size_t)Nn * Hh * 4, stream);
  (void)hipMemsetAsync(out_s_agg, 0, (size_t)Nn * Ss * 4, stream);
  (void)hipMemsetAsync(out_v_agg, 0, (size_t)3 * Nn * Vv * 4, stream);

  k_edge_logits<<<(Ee + 7) / 8, 256, 0, stream>>>(
      edge_index, q_s, k_s, q_v, k_v, edge_vec, W_edge, b_edge, logits, maxbuf);
  k_edge_expsum<<<(Ee * Hh + 255) / 256, 256, 0, stream>>>(edge_index, logits,
                                                           maxbuf, sumbuf);
  k_edge_scatter<<<(Ee + 7) / 8, 256, 0, stream>>>(
      edge_index, logits, sumbuf, v_s, v_v, out_s_agg, out_v_agg);

  // 4) output projections (WMMA)
  cast(out_s_agg, out_s_bf, Nn * Ss);
  cast(out_v_agg, out_v_bf, 3 * Nn * Vv);
  gemm(out_s_bf, Wos_bf, bout_s, out_s_fin, Nn, Ss, Ss);
  for (int i = 0; i < 3; ++i)
    gemm(out_v_bf + (size_t)i * Nn * Vv, Wov_bf, bout_v,
         out_v_fin + (size_t)i * Nn * Vv, Nn, Vv, Vv);

  // 5) residual + LayerNorm -> d_out[0 : N*S]; vector residual -> tail
  float* out_scalars = (float*)d_out;
  float* out_vectors = (float*)d_out + (size_t)Nn * Ss;
  k_layernorm<<<(Nn + 7) / 8, 256, 0, stream>>>(scalars, out_s_fin, ln_g, ln_b,
                                                out_scalars);
  k_add_vec<<<(Nn * Vv * 3 + 255) / 256, 256, 0, stream>>>(vectors, out_v_fin,
                                                           out_vectors);
}
